// LSTMWithSubnets_25864293056585
// MI455X (gfx1250) — matrix operand order repaired and checked by reading, not compiled, not run on hardware
//
#include <hip/hip_runtime.h>
#include <hip/hip_bf16.h>

// ---------------- problem constants ----------------
#define B_ROWS   131072
#define DY       31          // y feature count (D-1)
#define D_IN     32          // LSTM input dim
#define H        64          // LSTM hidden
#define GATES    256         // 4*H
#define S        512         // subnet width
#define NSTEPS   48
#define SUBIN    65
#define BN_EPS   1e-5f
#define KP       32          // layer-2 K panel width
#define NPANEL   (S / KP)    // 16

typedef __attribute__((ext_vector_type(16))) __bf16 v16bf;
typedef __attribute__((ext_vector_type(8)))  float  v8f;
typedef __attribute__((ext_vector_type(4)))  unsigned int u32x4;
typedef __attribute__((ext_vector_type(8)))  int  i32x8;
typedef __attribute__((ext_vector_type(4)))  int  i32x4;

union Frag { v16bf v; __bf16 e[16]; uint4 q[2]; };
union Acc8 { v8f v; float e[8]; };

#if __has_builtin(__builtin_amdgcn_tensor_load_to_lds)
#define HAVE_TDM 1
#else
#define HAVE_TDM 0
#endif

// ---------------- workspace layout (float indices), ~614 KB ----------------
#define WS_SUM    0
#define WS_SUMSQ  32
#define WS_MEAN   64
#define WS_RSTD   96
#define WS_IDX    128    // int (time_idx)
#define WS_BIASG  132    // 256 (bi2h + bh2h)
#define WS_B1     (WS_BIASG + GATES)
#define WS_B2     (WS_B1 + S)
#define WS_W3     (WS_B2 + S)
#define WS_B3     (WS_W3 + S)
#define WS_BF16F  1928   // 16B-aligned bf16 region
#define BF_WI2H   0                      // 256*32
#define BF_W1     (BF_WI2H + GATES*D_IN) // 512*64 (col 0 of W1 dropped: dummy_t==0)
#define BF_W2     (BF_W1 + S*H)          // 512*512

#define LOG2E 1.44269504088896340736f
// branch-free transcendentals: v_exp_f32 + v_rcp_f32, correct at +/-inf
__device__ __forceinline__ float sigm(float x) {
  return __builtin_amdgcn_rcpf(1.0f + __builtin_amdgcn_exp2f(-x * LOG2E));
}
__device__ __forceinline__ float tanh_fast(float x) {
  return fmaf(-2.0f, __builtin_amdgcn_rcpf(1.0f + __builtin_amdgcn_exp2f(2.0f * LOG2E * x)), 1.0f);
}

// A fragment: 16x32 bf16 tile from row-major src (leading dim ldk).
__device__ __forceinline__ v16bf load_a_frag(const __bf16* src, int ldk, int rM, int k0, int lane) {
  int r  = rM + (lane & 15);
  int kh = (lane >> 4) << 3;               // 0 or 8
  const __bf16* p = src + (size_t)r * ldk + k0 + kh;
  Frag f;
  f.q[0] = *(const uint4*)(p);
  f.q[1] = *(const uint4*)(p + 16);
  return f.v;
}
// B fragment: 32x16 bf16 tile from row-major weight [N][K].
__device__ __forceinline__ v16bf load_b_frag(const __bf16* w, int ldk, int n0, int k0, int lane) {
  int n = n0 + (lane & 15);
  int k = k0 + ((lane >> 4) << 3);         // 0 or 8
  const __bf16* p = w + (size_t)n * ldk + k;
  Frag f;
  f.q[0] = *(const uint4*)(p);
  f.q[1] = *(const uint4*)(p + 16);
  return f.v;
}

#if HAVE_TDM
// TDM: DMA a [512 x KP] bf16 tile (rows stride S in global) into LDS.
// 6-arg builtin (clang-23 arity): (v4u g0, v8i g1, v4i g2, v4i g3, v8i extra, i32 cpol)
__device__ __forceinline__ void tdm_load_panel(const __bf16* gsrc, unsigned lds_byte_off) {
  unsigned long long ga = (unsigned long long)(uintptr_t)gsrc;
  u32x4 g0 = { 1u,                                   // count = 1 valid descriptor
               lds_byte_off,                         // lds_addr
               (unsigned)ga,                         // global_addr[31:0]
               (unsigned)(ga >> 32) | (2u << 30) };  // global_addr[56:32] | type=2
  i32x8 g1 = { (int)(1u << 16),          // data_size = 1 (2 bytes), mask=0, no pad/iter
               (int)(512u << 16),        // tensor_dim0[15:0] = 512 at bits[63:48]
               (int)(512u << 16),        // tensor_dim0 hi=0, tensor_dim1[15:0]=512
               (int)((unsigned)KP << 16),// tile_dim0 = KP at bits[127:112]
               512,                      // tile_dim1 = 512 at bits[143:128]
               512,                      // tensor_dim0_stride[31:0] = 512
               0, 0 };                   // stride hi / tensor_dim1_stride = 0
  i32x4 g2 = { 0, 0, 0, 0 };
  i32x4 g3 = { 0, 0, 0, 0 };
  i32x8 g4 = { 0, 0, 0, 0, 0, 0, 0, 0 }; // unused trailing group (2D tile)
  __builtin_amdgcn_tensor_load_to_lds(g0, g1, g2, g3, g4, 0);
}
#endif

// ---------------- kernel 1: zero reduction buffers ----------------
__global__ void zero_stats_k(float* ws) {
  int i = threadIdx.x;
  if (i < 64) ws[WS_SUM + i] = 0.0f;
}

// ---------------- kernel 2: per-column sum / sumsq of y ----------------
__global__ void bn_reduce_k(const float* __restrict__ y, float* ws) {
  int lane = threadIdx.x & 31;
  int grp  = threadIdx.x >> 5;
  int r0   = blockIdx.x * 1024 + grp * 128;
  if (lane < DY) {
    const float* p = y + (size_t)r0 * DY + lane;
    float s = 0.0f, q = 0.0f;
    for (int i = 0; i < 128; ++i) { float v = p[(size_t)i * DY]; s += v; q += v * v; }
    atomicAdd(&ws[WS_SUM + lane], s);
    atomicAdd(&ws[WS_SUMSQ + lane], q);
  }
}

// ---------------- kernel 3: mean/rstd, time_idx, fused gate bias ----------------
__global__ void finalize_k(const float* __restrict__ t, const float* __restrict__ bi2h,
                           const float* __restrict__ bh2h, float* ws) {
  int i = threadIdx.x;
  if (i < DY) {
    float m = ws[WS_SUM + i] * (1.0f / (float)B_ROWS);
    float v = ws[WS_SUMSQ + i] * (1.0f / (float)B_ROWS) - m * m;
    ws[WS_MEAN + i] = m;
    ws[WS_RSTD + i] = rsqrtf(v + BN_EPS);
  }
  if (i < GATES) ws[WS_BIASG + i] = bi2h[i] + bh2h[i];
  if (i == 0) {
    int ti = (int)(t[0] * (float)NSTEPS);
    ti = ti > NSTEPS - 1 ? NSTEPS - 1 : (ti < 0 ? 0 : ti);
    ((int*)ws)[WS_IDX] = ti;
  }
}

// ---------------- kernel 4: convert selected-step weights to bf16 ----------------
__global__ void convert_k(const float* __restrict__ Wi2h, const float* __restrict__ W1,
                          const float* __restrict__ b1,  const float* __restrict__ W2,
                          const float* __restrict__ b2,  const float* __restrict__ W3,
                          const float* __restrict__ b3,  float* ws) {
  int idx = ((const int*)ws)[WS_IDX];
  __bf16* bfb = (__bf16*)(ws + WS_BF16F);
  int gid = blockIdx.x * blockDim.x + threadIdx.x;
  int stride = gridDim.x * blockDim.x;
  const float* W2s = W2 + (size_t)idx * S * S;
  for (int i = gid; i < S * S; i += stride) bfb[BF_W2 + i] = (__bf16)W2s[i];
  const float* W1s = W1 + (size_t)idx * S * SUBIN;
  for (int i = gid; i < S * H; i += stride) {
    int r = i >> 6, c = i & 63;
    bfb[BF_W1 + i] = (__bf16)W1s[r * SUBIN + 1 + c];
  }
  for (int i = gid; i < GATES * D_IN; i += stride) bfb[BF_WI2H + i] = (__bf16)Wi2h[i];
  for (int i = gid; i < S; i += stride) {
    ws[WS_B1 + i] = b1[idx * S + i];
    ws[WS_B2 + i] = b2[idx * S + i];
    ws[WS_W3 + i] = W3[idx * S + i];
  }
  if (gid == 0) ws[WS_B3] = b3[idx];
}

// ---------------- kernel 5: fused BN -> LSTM -> subnet, 64 rows / block ----------------
__launch_bounds__(256)
__global__ void fused_main_k(const float* __restrict__ t, const float* __restrict__ y,
                             const float* __restrict__ ws, float* __restrict__ out) {
  __shared__ __bf16 sU[64][D_IN];       // 4 KB
  __shared__ __bf16 sX[64][H];          // 8 KB
  __shared__ __bf16 sZ1[64][S];         // 64 KB
  __shared__ __bf16 sW2p[2][S][KP];     // 64 KB double-buffered W2 K-panels (TDM dest)
  __shared__ float  sBias[GATES];
  __shared__ float  sB1[S], sB2[S], sW3[S];
  __shared__ float  sMean[32], sRstd[32];
  __shared__ float  sOut[64];

  const int tid  = threadIdx.x;
  const int lane = tid & 31;
  const int wave = tid >> 5;
  const int rowBase = blockIdx.x * 64;
  const __bf16* bfb   = (const __bf16*)(ws + WS_BF16F);
  const __bf16* Wi2hb = bfb + BF_WI2H;
  const __bf16* W1b   = bfb + BF_W1;
  const __bf16* W2b   = bfb + BF_W2;

  // stage 0: parameters -> LDS
  for (int i = tid; i < GATES; i += 256) sBias[i] = ws[WS_BIASG + i];
  for (int i = tid; i < S; i += 256) {
    sB1[i] = ws[WS_B1 + i]; sB2[i] = ws[WS_B2 + i]; sW3[i] = ws[WS_W3 + i];
  }
  if (tid < DY) { sMean[tid] = ws[WS_MEAN + tid]; sRstd[tid] = ws[WS_RSTD + tid]; }
  if (tid < 64) sOut[tid] = ws[WS_B3];
#if HAVE_TDM
  // kick off the first W2 panel DMA early; nothing reads sW2p before layer 2
  if (wave == 0) tdm_load_panel(W2b, (unsigned)(uintptr_t)&sW2p[0][0][0]);
#endif
  __syncthreads();

  // stage 0b: u = [rel_t, bn(y)] in bf16
  for (int i = tid; i < 64 * D_IN; i += 256) {
    int r = i >> 5, c = i & 31;
    float v = (c == 0) ? t[rowBase + r]
                       : (y[(size_t)(rowBase + r) * DY + (c - 1)] - sMean[c - 1]) * sRstd[c - 1];
    sU[r][c] = (__bf16)v;
  }
  __syncthreads();

  // stage 1: LSTM gates (K=32, single WMMA per tile); i/g/o tiles per wave -> LSTM in regs
  for (int q = 0; q < 2; ++q) {
    int p  = wave * 2 + q;
    int rt = p >> 2, ct = p & 3;
    v16bf a  = load_a_frag(&sU[0][0], D_IN, rt * 16, 0, lane);
    v16bf bi = load_b_frag(Wi2hb, D_IN, 0 * H + ct * 16, 0, lane);
    v16bf bg = load_b_frag(Wi2hb, D_IN, 2 * H + ct * 16, 0, lane);
    v16bf bo = load_b_frag(Wi2hb, D_IN, 3 * H + ct * 16, 0, lane);
    int n = ct * 16 + (lane & 15);
    Acc8 ai, ag, ao;
    float vi = sBias[0 * H + n], vg = sBias[2 * H + n], vo = sBias[3 * H + n];
    for (int k = 0; k < 8; ++k) { ai.e[k] = vi; ag.e[k] = vg; ao.e[k] = vo; }
    ai.v = __builtin_amdgcn_wmma_f32_16x16x32_bf16(false, a, false, bi, (short)0, ai.v, false, false);
    ag.v = __builtin_amdgcn_wmma_f32_16x16x32_bf16(false, a, false, bg, (short)0, ag.v, false, false);
    ao.v = __builtin_amdgcn_wmma_f32_16x16x32_bf16(false, a, false, bo, (short)0, ao.v, false, false);
    int m0 = rt * 16 + ((lane >> 4) << 3);
    for (int k = 0; k < 8; ++k) {
      float c_ = sigm(ai.e[k]) * tanh_fast(ag.e[k]);   // c0 = 0
      float h_ = sigm(ao.e[k]) * tanh_fast(c_);
      sX[m0 + k][n] = (__bf16)tanh_fast(h_);
    }
  }
  __syncthreads();

  // stage 2: z1 = tanh(x @ W1b^T + b1), K=64 (2 WMMA steps), 128 tiles / 8 waves
  for (int j = 0; j < 16; ++j) {
    int jj = wave * 16 + j;
    int rt = jj >> 5, nt = jj & 31;
    int n = nt * 16 + (lane & 15);
    Acc8 acc; float bv = sB1[n];
    for (int k = 0; k < 8; ++k) acc.e[k] = bv;
    for (int k0 = 0; k0 < H; k0 += 32) {
      v16bf a = load_a_frag(&sX[0][0], H, rt * 16, k0, lane);
      v16bf b = load_b_frag(W1b, H, nt * 16, k0, lane);
      acc.v = __builtin_amdgcn_wmma_f32_16x16x32_bf16(false, a, false, b, (short)0, acc.v, false, false);
    }
    int m0 = rt * 16 + ((lane >> 4) << 3);
    for (int k = 0; k < 8; ++k) sZ1[m0 + k][n] = (__bf16)tanh_fast(acc.e[k]);
  }
  __syncthreads();

  // stage 3: z2 = tanh(z1 @ W2^T + b2) with W2 streamed through LDS by the TDM,
  // double-buffered KP-wide K panels; W3 projection folded into the epilogue.
  {
    const int rt = wave >> 1;              // fixed row-tile per wave
    const int ntBase = (wave & 1) * 16;    // 16 N-tiles per wave
    Acc8 acc[16];
#pragma unroll
    for (int j = 0; j < 16; ++j) {
      float bv = sB2[(ntBase + j) * 16 + (lane & 15)];
      for (int k = 0; k < 8; ++k) acc[j].e[k] = bv;
    }
    for (int p = 0; p < NPANEL; ++p) {
#if HAVE_TDM
      if (wave == 0) {
        if (p + 1 < NPANEL) {
          tdm_load_panel(W2b + (size_t)(p + 1) * KP,
                         (unsigned)(uintptr_t)&sW2p[(p + 1) & 1][0][0]);
          __builtin_amdgcn_s_wait_tensorcnt(1);   // panel p complete (in-order)
        } else {
          __builtin_amdgcn_s_wait_tensorcnt(0);
        }
      }
#else
      // fallback: cooperative copy of panel p
      for (int i = tid; i < (S * KP) / 8; i += 256) {
        int n = i >> 2;              // KP/8 = 4 16B-chunks per row
        int c8 = (i & 3) << 3;
        *(uint4*)&sW2p[p & 1][n][c8] = *(const uint4*)(W2b + (size_t)n * S + p * KP + c8);
      }
#endif
      __syncthreads();
      const __bf16* panel = &sW2p[p & 1][0][0];
      v16bf a = load_a_frag(&sZ1[0][0], S, rt * 16, p * KP, lane);
#pragma unroll
      for (int j = 0; j < 16; ++j) {
        v16bf b = load_b_frag(panel, KP, (ntBase + j) * 16, 0, lane);
        acc[j].v = __builtin_amdgcn_wmma_f32_16x16x32_bf16(false, a, false, b, (short)0, acc[j].v, false, false);
      }
      __syncthreads();               // all waves done reading before buffer reuse
    }
#pragma unroll
    for (int j = 0; j < 16; ++j) {
      int n = (ntBase + j) * 16 + (lane & 15);
      float w3n = sW3[n];
      int m0 = rt * 16 + ((lane >> 4) << 3);
      for (int k = 0; k < 8; ++k) atomicAdd(&sOut[m0 + k], tanh_fast(acc[j].e[k]) * w3n);
    }
  }
  __syncthreads();

  if (tid < 64) out[rowBase + tid] = sOut[tid];
}

extern "C" void kernel_launch(void* const* d_in, const int* in_sizes, int n_in,
                              void* d_out, int out_size, void* d_ws, size_t ws_size,
                              hipStream_t stream) {
  const float* t    = (const float*)d_in[0];
  const float* y    = (const float*)d_in[1];
  const float* Wi2h = (const float*)d_in[2];
  const float* bi2h = (const float*)d_in[3];
  /* d_in[4] = Wh2h unused: h0 == 0 */
  const float* bh2h = (const float*)d_in[5];
  const float* W1   = (const float*)d_in[6];
  const float* b1   = (const float*)d_in[7];
  const float* W2   = (const float*)d_in[8];
  const float* b2   = (const float*)d_in[9];
  const float* W3   = (const float*)d_in[10];
  const float* b3   = (const float*)d_in[11];
  float* ws  = (float*)d_ws;
  float* out = (float*)d_out;

  zero_stats_k<<<1, 64, 0, stream>>>(ws);
  bn_reduce_k<<<128, 256, 0, stream>>>(y, ws);
  finalize_k<<<1, 256, 0, stream>>>(t, bi2h, bh2h, ws);
  convert_k<<<512, 256, 0, stream>>>(Wi2h, W1, b1, W2, b2, W3, b3, ws);
  fused_main_k<<<B_ROWS / 64, 256, 0, stream>>>(t, y, ws, out);
}
